// MFWRelativeSelfMultiheadAttn_26989574488824
// MI455X (gfx1250) — matrix operand order repaired and checked by reading, not compiled, not run on hardware
//
#include <hip/hip_runtime.h>
#include <math.h>

#define E_C    1024
#define H_C    16
#define T_C    1024
#define B_C    4
#define D_C    64
#define RANK_C 4

typedef __bf16 bf16_t;
typedef __attribute__((ext_vector_type(16))) __bf16 v16bf;
typedef __attribute__((ext_vector_type(8)))  __bf16 v8bf;
typedef __attribute__((ext_vector_type(8)))  float  v8f;

static __device__ __forceinline__ v8f wmma_bf(v16bf a, v16bf b, v8f c) {
  return __builtin_amdgcn_wmma_f32_16x16x32_bf16(false, a, false, b, (short)0, c,
                                                 false, false);
}

// A operand (16x32 bf16): lane holds row (lane&15); contiguous K halves
// (matching frag_b_s): half=0 -> K 0..15 ; half=1 -> K 16..31.
static __device__ __forceinline__ v16bf frag_a(const bf16_t* row, int half) {
  v16bf a;
  const bf16_t* p = row + half * 16;
#pragma unroll
  for (int t = 0; t < 8; ++t) { a[t] = p[t]; a[8 + t] = p[8 + t]; }
  return a;
}

// B operand (32x16 bf16): lane holds column (lane&15); 16 consecutive K values
// starting at half*16, separated by `stride` elements.
static __device__ __forceinline__ v16bf frag_b_s(const bf16_t* colbase, int stride,
                                                 int half) {
  v16bf b;
  const bf16_t* p = colbase + (size_t)half * 16 * stride;
#pragma unroll
  for (int e = 0; e < 16; ++e) b[e] = p[(size_t)e * stride];
  return b;
}

// CDNA5 async global->LDS copy (16B per lane), tracked by ASYNCcnt.
static __device__ __forceinline__ void async_copy_b128(unsigned lds_off,
                                                       const void* gaddr) {
  asm volatile("global_load_async_to_lds_b128 %0, %1, off"
               :
               : "v"(lds_off), "v"(gaddr)
               : "memory");
}
static __device__ __forceinline__ void wait_asynccnt0() {
  asm volatile("s_wait_asynccnt 0x0" ::: "memory");
}

// ---------------- elementwise helpers ----------------

__global__ void k_weight_update(const float* __restrict__ W,
                                const float* __restrict__ r,
                                const float* __restrict__ s,
                                const int* __restrict__ indices,
                                bf16_t* __restrict__ out, int F) {
  size_t idx = (size_t)blockIdx.x * blockDim.x + threadIdx.x;
  int e = (int)(idx / F);
  int f = (int)(idx % F);
  int lang = indices[0];
  const float* rb = r + (size_t)lang * RANK_C * E_C;
  const float* sb = s + (size_t)lang * RANK_C * F;
  float acc = W[idx];
#pragma unroll
  for (int k = 0; k < RANK_C; ++k)
    acc += rb[(size_t)k * E_C + e] * sb[(size_t)k * F + f];
  out[idx] = (bf16_t)acc;
}

__global__ void k_cast_bf16(const float* __restrict__ in, bf16_t* __restrict__ out) {
  size_t idx = (size_t)blockIdx.x * blockDim.x + threadIdx.x;
  out[idx] = (bf16_t)in[idx];
}

// qkv [T,B,3E] fp32 (bias already added) -> per-(b,h) bf16 q+rw, q+rr, k, v in
// [B,H,T,D] layout.
__global__ void k_qkv_post(const float* __restrict__ qkv,
                           const float* __restrict__ rwb,
                           const float* __restrict__ rrb,
                           bf16_t* __restrict__ qw, bf16_t* __restrict__ qr,
                           bf16_t* __restrict__ kk, bf16_t* __restrict__ vv) {
  size_t idx = (size_t)blockIdx.x * blockDim.x + threadIdx.x;  // [B][H][T][D]
  int d = (int)(idx % D_C);
  size_t r1 = idx / D_C;
  int t = (int)(r1 % T_C);
  size_t r2 = r1 / T_C;
  int h = (int)(r2 % H_C);
  int b = (int)(r2 / H_C);
  int hd = h * D_C + d;
  size_t base = ((size_t)t * B_C + b) * (3 * E_C);
  float q = qkv[base + hd];
  qw[idx] = (bf16_t)(q + rwb[hd]);
  qr[idx] = (bf16_t)(q + rrb[hd]);
  kk[idx] = (bf16_t)qkv[base + E_C + hd];
  vv[idx] = (bf16_t)qkv[base + 2 * E_C + hd];
}

// r_head_k [T,E]=[T,H,D] fp32 -> [H,T,D] bf16
__global__ void k_rk_post(const float* __restrict__ rh, bf16_t* __restrict__ rk) {
  size_t idx = (size_t)blockIdx.x * blockDim.x + threadIdx.x;  // [H][T][D]
  int d = (int)(idx % D_C);
  size_t r1 = idx / D_C;
  int t = (int)(r1 % T_C);
  int h = (int)(r1 / T_C);
  rk[idx] = (bf16_t)rh[(size_t)t * E_C + h * D_C + d];
}

// ---- WMMA GEMM: C[M,N] = A[M,K](bf16) @ B[K,N](bf16) + bias[N], N/K static,
// ---- double-buffered LDS with async A staging -------------------------------

template <int N, int K>
__global__ __launch_bounds__(128) void k_gemm(const bf16_t* __restrict__ A,
                                              const bf16_t* __restrict__ B,
                                              const float* __restrict__ bias,
                                              float* __restrict__ C, int M) {
  // Single static LDS object at offset 0; As is its first member, so the
  // async-copy LDS byte offsets below are plain offsets into As.
  struct __align__(16) Smem {
    bf16_t As[2][64][32];  // buffer b at byte offset b*4096
    bf16_t Bt[2][64][34];  // K-transposed, padded
  };
  __shared__ Smem sm;
  const int tid = threadIdx.x;
  const int lane = tid & 31;
  const int wv = tid >> 5;
  const int half = (lane >> 4) & 1;
  const int l16 = lane & 15;
  const int m0 = blockIdx.y * 64;
  const int n0 = blockIdx.x * 64;
  const int wm = (wv >> 1) * 32;
  const int wn = (wv & 1) * 32;
  const int arow = tid >> 1;       // 0..63
  const int ahk = (tid & 1) * 16;  // 0 or 16
  const int bkk = tid >> 2;        // 0..31
  const int bnb = (tid & 3) * 16;  // 0,16,32,48

  bf16_t breg[16];
  auto stage_a = [&](int k0, int buf) {
    const bf16_t* src = A + (size_t)(m0 + arow) * K + k0 + ahk;
    unsigned loff = (unsigned)(buf * 4096 + arow * 64 + ahk * 2);
    async_copy_b128(loff, src);
    async_copy_b128(loff + 16, src + 8);
  };
  auto load_b = [&](int k0) {
    const bf16_t* src = B + (size_t)(k0 + bkk) * N + n0 + bnb;
#pragma unroll
    for (int i = 0; i < 16; ++i) breg[i] = src[i];
  };
  auto store_b = [&](int buf) {
#pragma unroll
    for (int i = 0; i < 16; ++i) sm.Bt[buf][bnb + i][bkk] = breg[i];
  };

  // prologue: fill buffer 0
  stage_a(0, 0);
  load_b(0);
  store_b(0);
  wait_asynccnt0();
  __syncthreads();

  v8f c[2][2] = {};
  for (int k0 = 0; k0 < K; k0 += 32) {
    const int cur = (k0 >> 5) & 1;
    const int nxt = cur ^ 1;
    const bool more = (k0 + 32) < K;
    if (more) {  // start staging next tile while computing this one
      stage_a(k0 + 32, nxt);
      load_b(k0 + 32);
    }
    if (k0 + 64 < K) {
      __builtin_prefetch(A + (size_t)(m0 + arow) * K + k0 + 64 + ahk, 0, 1);
      __builtin_prefetch(B + (size_t)(k0 + 64 + bkk) * N + n0 + bnb, 0, 1);
    }
    v16bf af[2], bfv[2];
#pragma unroll
    for (int mi = 0; mi < 2; ++mi)
      af[mi] = frag_a(&sm.As[cur][wm + mi * 16 + l16][0], half);
#pragma unroll
    for (int ni = 0; ni < 2; ++ni)
      bfv[ni] = frag_b_s(&sm.Bt[cur][wn + ni * 16 + l16][0], 1, half);
#pragma unroll
    for (int mi = 0; mi < 2; ++mi)
#pragma unroll
      for (int ni = 0; ni < 2; ++ni) c[mi][ni] = wmma_bf(af[mi], bfv[ni], c[mi][ni]);
    if (more) store_b(nxt);
    wait_asynccnt0();
    __syncthreads();
  }
#pragma unroll
  for (int mi = 0; mi < 2; ++mi)
#pragma unroll
    for (int ni = 0; ni < 2; ++ni) {
      int col = n0 + wn + ni * 16 + l16;
      float bz = bias[col];
#pragma unroll
      for (int v = 0; v < 8; ++v) {
        int row = m0 + wm + mi * 16 + v + 8 * half;
        C[(size_t)row * N + col] = c[mi][ni][v] + bz;
      }
    }
}

// ---------------- attention: one wave per (b, h, 16-query block) ----------------

__global__ __launch_bounds__(32) void k_attn(
    const bf16_t* __restrict__ qw, const bf16_t* __restrict__ qr,
    const bf16_t* __restrict__ kmat, const bf16_t* __restrict__ vmat,
    const bf16_t* __restrict__ rk, const unsigned char* __restrict__ mask,
    bf16_t* __restrict__ ctx) {
  // bd stored transposed: bd[col][row], row-stride 24 bf16 (48B) so each lane's
  // 8 row-values are one aligned 16B ds_store_b128.
  struct __align__(16) AttnSmem {
    bf16_t bd[T_C * 24];  // 48KB
    bf16_t p[16 * 32];    // probability tile for C->A transpose
  };
  __shared__ AttnSmem sm;
  const int lane = threadIdx.x;
  const int half = lane >> 4;
  const int l16 = lane & 15;
  const int i0 = blockIdx.x * 16;
  const int h = blockIdx.y;
  const int b = blockIdx.z;
  const size_t bh = ((size_t)b * H_C + h) * (size_t)T_C * D_C;
  const bf16_t* qw_b = qw + bh;
  const bf16_t* qr_b = qr + bh;
  const bf16_t* k_b = kmat + bh;
  const bf16_t* v_b = vmat + bh;
  const bf16_t* rk_h = rk + (size_t)h * T_C * D_C;

  // Phase 1: bd[i0 .. i0+16][0..T) = (q + r_r_bias) @ r_head_k^T via WMMA.
  // mt=0 covers rows i0..i0+15 (store all 8 C rows), mt=1 contributes only
  // row i0+16 (C row 0 -> v=0, half=0): single predicated store.
#pragma unroll
  for (int mt = 0; mt < 2; ++mt) {
    int row = i0 + mt * 16 + l16;
    if (row > T_C - 1) row = T_C - 1;  // padded rows (never read back)
    v16bf aq0 = frag_a(qr_b + (size_t)row * D_C, half);
    v16bf aq1 = frag_a(qr_b + (size_t)row * D_C + 32, half);
    v16bf b0 = frag_b_s(rk_h + (size_t)l16 * D_C, 1, half);
    v16bf b1 = frag_b_s(rk_h + (size_t)l16 * D_C + 32, 1, half);
    for (int n0 = 0; n0 < T_C; n0 += 16) {
      v16bf nb0 = b0, nb1 = b1;
      if (n0 + 16 < T_C) {  // pipeline next column tile
        const bf16_t* nr = rk_h + (size_t)(n0 + 16 + l16) * D_C;
        nb0 = frag_b_s(nr, 1, half);
        nb1 = frag_b_s(nr + 32, 1, half);
      }
      v8f acc = {};
      acc = wmma_bf(aq0, b0, acc);
      acc = wmma_bf(aq1, b1, acc);
      if (mt == 0) {
        v8bf pk;
#pragma unroll
        for (int v = 0; v < 8; ++v) pk[v] = (bf16_t)acc[v];
        *reinterpret_cast<v8bf*>(&sm.bd[(size_t)(n0 + l16) * 24 + half * 8]) = pk;
      } else {
        if (half == 0) sm.bd[(size_t)(n0 + l16) * 24 + 16] = (bf16_t)acc[0];
      }
      b0 = nb0;
      b1 = nb1;
    }
  }
  __syncthreads();

  // Phase 2: flash loop over key tiles of 32
  float mrow[8], lrow[8], alpha[8];
  v8f o[4] = {};
#pragma unroll
  for (int v = 0; v < 8; ++v) { mrow[v] = -INFINITY; lrow[v] = 0.f; }

  v16bf aqw0 = frag_a(qw_b + (size_t)(i0 + l16) * D_C, half);
  v16bf aqw1 = frag_a(qw_b + (size_t)(i0 + l16) * D_C + 32, half);

  v16bf kb[2][2];
  {
    const bf16_t* kr0 = k_b + (size_t)l16 * D_C;
    const bf16_t* kr1 = k_b + (size_t)(16 + l16) * D_C;
    kb[0][0] = frag_b_s(kr0, 1, half);
    kb[0][1] = frag_b_s(kr0 + 32, 1, half);
    kb[1][0] = frag_b_s(kr1, 1, half);
    kb[1][1] = frag_b_s(kr1 + 32, 1, half);
  }

  for (int j0 = 0; j0 < T_C; j0 += 32) {
    v16bf nk[2][2];
#pragma unroll
    for (int f = 0; f < 2; ++f) {
      nk[f][0] = kb[f][0];
      nk[f][1] = kb[f][1];
    }
    if (j0 + 32 < T_C) {  // pipeline next K tile
      const bf16_t* kr0 = k_b + (size_t)(j0 + 32 + l16) * D_C;
      const bf16_t* kr1 = k_b + (size_t)(j0 + 48 + l16) * D_C;
      nk[0][0] = frag_b_s(kr0, 1, half);
      nk[0][1] = frag_b_s(kr0 + 32, 1, half);
      nk[1][0] = frag_b_s(kr1, 1, half);
      nk[1][1] = frag_b_s(kr1 + 32, 1, half);
      __builtin_prefetch(v_b + (size_t)(j0 + 32 + lane) * D_C, 0, 1);
    }
    v8f s[2] = {};
#pragma unroll
    for (int f = 0; f < 2; ++f) {
      s[f] = wmma_bf(aqw0, kb[f][0], s[f]);
      s[f] = wmma_bf(aqw1, kb[f][1], s[f]);
    }
    // V fragments for this tile, issued early to overlap with softmax math
    v16bf vb[4];
#pragma unroll
    for (int n = 0; n < 4; ++n)
      vb[n] = frag_b_s(v_b + (size_t)j0 * D_C + n * 16 + l16, D_C, half);
    // rel-shifted bd gather + scale + mask
#pragma unroll
    for (int f = 0; f < 2; ++f) {
      int j = j0 + f * 16 + l16;
      bool msk = mask[(size_t)b * T_C + j] != 0;
#pragma unroll
      for (int v = 0; v < 8; ++v) {
        int i = i0 + v + 8 * half;
        int p = (i + 1) * T_C + j;
        int a_ = p / (T_C + 1);
        int c_ = p - a_ * (T_C + 1);
        float bdv = (c_ == 0) ? 0.f : (float)sm.bd[(size_t)(c_ - 1) * 24 + (a_ - i0)];
        float sc = (s[f][v] + bdv) * 0.125f;
        s[f][v] = msk ? -1e9f : sc;
      }
    }
    // online softmax (row groups = 16 lanes, matching C-fragment row mapping)
#pragma unroll
    for (int v = 0; v < 8; ++v) {
      float rm = fmaxf(s[0][v], s[1][v]);
#pragma unroll
      for (int off = 8; off >= 1; off >>= 1) rm = fmaxf(rm, __shfl_xor(rm, off, 16));
      float mn = fmaxf(mrow[v], rm);
      alpha[v] = __expf(mrow[v] - mn);
      float p0 = __expf(s[0][v] - mn);
      float p1 = __expf(s[1][v] - mn);
      s[0][v] = p0;
      s[1][v] = p1;
      float rs = p0 + p1;
#pragma unroll
      for (int off = 8; off >= 1; off >>= 1) rs += __shfl_xor(rs, off, 16);
      lrow[v] = lrow[v] * alpha[v] + rs;
      mrow[v] = mn;
    }
#pragma unroll
    for (int n = 0; n < 4; ++n)
#pragma unroll
      for (int v = 0; v < 8; ++v) o[n][v] *= alpha[v];
    // C-layout -> A-layout transpose through LDS, then PV
    __syncthreads();
#pragma unroll
    for (int f = 0; f < 2; ++f)
#pragma unroll
      for (int v = 0; v < 8; ++v)
        sm.p[(v + 8 * half) * 32 + f * 16 + l16] = (bf16_t)s[f][v];
    __syncthreads();
    v16bf ap = frag_a(&sm.p[l16 * 32], half);
#pragma unroll
    for (int n = 0; n < 4; ++n) o[n] = wmma_bf(ap, vb[n], o[n]);
#pragma unroll
    for (int f = 0; f < 2; ++f) {
      kb[f][0] = nk[f][0];
      kb[f][1] = nk[f][1];
    }
  }
  // normalize + write ctx [T,B,E] bf16
#pragma unroll
  for (int n = 0; n < 4; ++n)
#pragma unroll
    for (int v = 0; v < 8; ++v) {
      int i = i0 + v + 8 * half;
      int d = n * 16 + l16;
      float val = o[n][v] / lrow[v];
      ctx[((size_t)i * B_C + b) * E_C + h * D_C + d] = (bf16_t)val;
    }
}

// ---------------- host orchestration ----------------

extern "C" void kernel_launch(void* const* d_in, const int* in_sizes, int n_in,
                              void* d_out, int out_size, void* d_ws, size_t ws_size,
                              hipStream_t stream) {
  (void)in_sizes; (void)n_in; (void)out_size; (void)ws_size;
  const float* input = (const float*)d_in[0];
  const float* pos = (const float*)d_in[1];
  const int* indices = (const int*)d_in[2];
  const unsigned char* mask = (const unsigned char*)d_in[3];
  const float* w_in = (const float*)d_in[4];
  const float* w_pos = (const float*)d_in[5];
  const float* w_out = (const float*)d_in[6];
  const float* b_in = (const float*)d_in[7];
  const float* b_pos = (const float*)d_in[8];
  const float* b_out = (const float*)d_in[9];
  const float* r_i = (const float*)d_in[10];
  const float* s_i = (const float*)d_in[11];
  const float* r_p = (const float*)d_in[12];
  const float* s_p = (const float*)d_in[13];
  const float* r_o = (const float*)d_in[14];
  const float* s_o = (const float*)d_in[15];
  const float* rwb = (const float*)d_in[16];
  const float* rrb = (const float*)d_in[17];
  float* out = (float*)d_out;

  char* ws = (char*)d_ws;
  size_t off = 0;
  auto carve = [&](size_t bytes) -> char* {
    char* p = ws + off;
    off += (bytes + 255) & ~(size_t)255;
    return p;
  };
  bf16_t* w_in_bf = (bf16_t*)carve((size_t)E_C * 3 * E_C * 2);
  bf16_t* w_pos_bf = (bf16_t*)carve((size_t)E_C * E_C * 2);
  bf16_t* w_out_bf = (bf16_t*)carve((size_t)E_C * E_C * 2);
  bf16_t* in_bf = (bf16_t*)carve((size_t)T_C * B_C * E_C * 2);
  bf16_t* pos_bf = (bf16_t*)carve((size_t)T_C * E_C * 2);
  float* qkv_f = (float*)carve((size_t)T_C * B_C * 3 * E_C * 4);
  float* rh_f = (float*)carve((size_t)T_C * E_C * 4);
  bf16_t* qw = (bf16_t*)carve((size_t)B_C * H_C * T_C * D_C * 2);
  bf16_t* qr = (bf16_t*)carve((size_t)B_C * H_C * T_C * D_C * 2);
  bf16_t* kk = (bf16_t*)carve((size_t)B_C * H_C * T_C * D_C * 2);
  bf16_t* vv = (bf16_t*)carve((size_t)B_C * H_C * T_C * D_C * 2);
  bf16_t* rk_bf = (bf16_t*)carve((size_t)H_C * T_C * D_C * 2);
  bf16_t* ctx_bf = (bf16_t*)carve((size_t)T_C * B_C * E_C * 2);

  // 1) low-rank weight updates -> bf16
  k_weight_update<<<(E_C * 3 * E_C) / 256, 256, 0, stream>>>(w_in, r_i, s_i, indices,
                                                             w_in_bf, 3 * E_C);
  k_weight_update<<<(E_C * E_C) / 256, 256, 0, stream>>>(w_pos, r_p, s_p, indices,
                                                         w_pos_bf, E_C);
  k_weight_update<<<(E_C * E_C) / 256, 256, 0, stream>>>(w_out, r_o, s_o, indices,
                                                         w_out_bf, E_C);
  // 2) activation casts
  k_cast_bf16<<<((size_t)T_C * B_C * E_C) / 256, 256, 0, stream>>>(input, in_bf);
  k_cast_bf16<<<((size_t)T_C * E_C) / 256, 256, 0, stream>>>(pos, pos_bf);
  // 3) projections (WMMA, double-buffered async staging)
  k_gemm<3 * E_C, E_C><<<dim3(3 * E_C / 64, (T_C * B_C) / 64), 128, 0, stream>>>(
      in_bf, w_in_bf, b_in, qkv_f, T_C * B_C);
  k_gemm<E_C, E_C><<<dim3(E_C / 64, T_C / 64), 128, 0, stream>>>(
      pos_bf, w_pos_bf, b_pos, rh_f, T_C);
  // 4) layout/bias post-processing
  k_qkv_post<<<((size_t)B_C * H_C * T_C * D_C) / 256, 256, 0, stream>>>(
      qkv_f, rwb, rrb, qw, qr, kk, vv);
  k_rk_post<<<((size_t)H_C * T_C * D_C) / 256, 256, 0, stream>>>(rh_f, rk_bf);
  // 5) attention with rel-shift (WMMA flash loop)
  k_attn<<<dim3(T_C / 16, H_C, B_C), 32, 0, stream>>>(qw, qr, kk, vv, rk_bf, mask,
                                                      ctx_bf);
  // 6) output projection -> fp32 d_out (WMMA)
  k_gemm<E_C, E_C><<<dim3(E_C / 64, (T_C * B_C) / 64), 128, 0, stream>>>(
      ctx_bf, w_out_bf, b_out, out, T_C * B_C);
}
